// CubeRecommender_4595615007501
// MI455X (gfx1250) — compile-verified, matrix operand order checked by reading, not run on hardware
//
#include <hip/hip_runtime.h>
#include <hip/hip_bf16.h>
#include <stdint.h>

typedef __attribute__((ext_vector_type(16))) _Float16 v16h;
typedef __attribute__((ext_vector_type(8)))  _Float16 v8h;
typedef __attribute__((ext_vector_type(4)))  _Float16 v4h;
typedef __attribute__((ext_vector_type(8)))  float    v8f;
typedef __attribute__((ext_vector_type(4)))  int      v4i;

#define NUM_CARDS 30000
#define N_GALLERY 29999
#define BATCH 256
#define CUBE 360
#define CDIM 512
#define HID 512
#define EDIM 256

// ---- async global->LDS copy (CDNA5 GLOBAL_LOAD_ASYNC_TO_LDS_B128) ----------
// Probe result (round 2 stderr): builtin exists, 4 args, param0 is an int4
// pointer in address space 1 (printed as "int __vector(4) __device__ *").
#if __has_builtin(__builtin_amdgcn_global_load_async_to_lds_b128)
#define USE_ASYNC_LDS 1
#endif

__device__ __forceinline__ void cp16_g2l(const _Float16* g, _Float16* l) {
#if defined(USE_ASYNC_LDS)
  __builtin_amdgcn_global_load_async_to_lds_b128(
      (__attribute__((address_space(1))) v4i*)(uintptr_t)g,
      (__attribute__((address_space(3))) v4i*)l, 0, 0);
#else
  *(uint4*)l = *(const uint4*)g;
#endif
}
__device__ __forceinline__ void cp_wait() {
#if defined(USE_ASYNC_LDS)
#if __has_builtin(__builtin_amdgcn_s_wait_asynccnt)
  __builtin_amdgcn_s_wait_asynccnt(0);
#else
  asm volatile("s_wait_asynccnt 0" ::: "memory");
#endif
#endif
}

// ---- fragment loader -------------------------------------------------------
// 16-bit A 16x32 layout (CDNA5 ISA 7.12.2): lane l holds row M=l%16.
// lanes 0-15:  VGPR0..3 = K 0..7,  VGPR4..7 = K 16..23
// lanes 16-31: VGPR0..3 = K 8..15, VGPR4..7 = K 24..31
// B 32x16 mirrors this with N=l%16, so the same loader serves row-major A rows
// and row-major B^T rows (columns of W).
__device__ __forceinline__ v16h frag16(const _Float16* p, int k0, int kbase) {
  v8h lo = *(const v8h*)(p + k0 + kbase);
  v8h hi = *(const v8h*)(p + k0 + 16 + kbase);
  return __builtin_shufflevector(lo, hi, 0,1,2,3,4,5,6,7,8,9,10,11,12,13,14,15);
}

__device__ __forceinline__ v8f wmma_f16(v16h a, v16h b, v8f c) {
  return __builtin_amdgcn_wmma_f32_16x16x32_f16(false, a, false, b, (short)0, c,
                                                false, false);
}

// K=512 tile GEMM, B fragments pre-hoisted into registers, A one-ahead pipelined.
__device__ __forceinline__ v8f gemm_k512_regB(const _Float16* arow,
                                              const v16h* bf, int kbase) {
  v16h a = frag16(arow, 0, kbase);
  v8f c = {};
#pragma unroll
  for (int ks = 0; ks < 16; ++ks) {
    v16h an = (ks < 15) ? frag16(arow, (ks + 1) * 32, kbase) : a;
    c = wmma_f16(a, bf[ks], c);
    a = an;
  }
  return c;
}

// K=512 tile GEMM, B loaded inline (used when B tiles are used only once).
__device__ __forceinline__ v8f gemm_k512_ldB(const _Float16* arow,
                                             const _Float16* Wrow, int kbase) {
  v16h a = frag16(arow, 0, kbase);
  v8f c = {};
#pragma unroll
  for (int ks = 0; ks < 16; ++ks) {
    v16h b = frag16(Wrow, ks * 32, kbase);
    v16h an = (ks < 15) ? frag16(arow, (ks + 1) * 32, kbase) : a;
    c = wmma_f16(a, b, c);
    a = an;
  }
  return c;
}

// ---- prep: f32 -> f16 row-major copy of the card gallery -------------------
__global__ __launch_bounds__(256) void k_f32_to_f16(
    const float* __restrict__ in, _Float16* __restrict__ out, int n4) {
  int i = blockIdx.x * blockDim.x + threadIdx.x;
  if (i < n4) {
    float4 v = ((const float4*)in)[i];
    v4h o;
    o[0] = (_Float16)v.x; o[1] = (_Float16)v.y;
    o[2] = (_Float16)v.z; o[3] = (_Float16)v.w;
    ((v4h*)out)[i] = o;
  }
}

// ---- prep: weight transpose + f32 -> f16 ----------------------------------
__global__ __launch_bounds__(256) void k_transpose_h(
    const float* __restrict__ in, _Float16* __restrict__ out, int K, int N) {
  int i = blockIdx.x * blockDim.x + threadIdx.x;
  if (i < K * N) {
    int k = i / N, n = i - k * N;
    out[(size_t)n * K + k] = (_Float16)in[i];
  }
}

// ---- stage 1: gather + layer1 + relu + mean pool ---------------------------
// grid (256 batches, 4 n-slices of 128), 256 threads (8 waves x 16 cols).
// Ping-pong LDS staging with async global->LDS copies overlapped with WMMA.
__device__ __forceinline__ void stage_embed_tile(
    const int* __restrict__ idx, const _Float16* __restrict__ cardsH,
    _Float16* buf, int b, int mt, int tid) {
  for (int j = tid; j < 16 * 64; j += 256) {   // 16 rows x 64 uint4
    int r = j >> 6, kq = j & 63;
    int s = mt * 16 + r; s = s < CUBE ? s : CUBE - 1;
    int card = idx[b * CUBE + s];
    cp16_g2l(cardsH + (size_t)card * CDIM + kq * 8, buf + r * 520 + kq * 8);
  }
}

__global__ __launch_bounds__(256) void k_embed_pool(
    const int* __restrict__ idx, const _Float16* __restrict__ cardsH,
    const _Float16* __restrict__ W1T, const float* __restrict__ b1,
    float* __restrict__ pooled) {
  __shared__ __align__(16) _Float16 AhA[16 * 520];
  __shared__ __align__(16) _Float16 AhB[16 * 520];
  const int b = blockIdx.x;
  const int slice = blockIdx.y;
  const int tid = threadIdx.x;
  const int wave = tid >> 5, lane = tid & 31;
  const int lrow = lane & 15;
  const int kbase = (lane < 16) ? 0 : 8;
  const int ncol = slice * 128 + wave * 16 + lrow;
  const float bias_n = b1[ncol];
  // hoist this wave's 16 B fragments (16 cols x K=512) into registers
  const _Float16* Wrow = W1T + (size_t)ncol * CDIM;
  v16h bf[16];
#pragma unroll
  for (int ks = 0; ks < 16; ++ks) bf[ks] = frag16(Wrow, ks * 32, kbase);

  float colsum = 0.f;
  stage_embed_tile(idx, cardsH, AhA, b, 0, tid);
  cp_wait();
  __syncthreads();
  for (int mt = 0; mt < 23; ++mt) {            // ceil(360/16)
    _Float16* cur = (mt & 1) ? AhB : AhA;
    _Float16* nxt = (mt & 1) ? AhA : AhB;
    if (mt + 1 < 23) stage_embed_tile(idx, cardsH, nxt, b, mt + 1, tid);
    v8f c = gemm_k512_regB(cur + lrow * 520, bf, kbase);
    const int rbase = mt * 16 + ((lane < 16) ? 0 : 8);
#pragma unroll
    for (int r = 0; r < 8; ++r) {
      float h = c[r] + bias_n;
      h = h > 0.f ? h : 0.f;
      colsum += (rbase + r < CUBE) ? h : 0.f;  // mask padded rows
    }
    cp_wait();
    __syncthreads();
  }
  colsum += __shfl_xor(colsum, 16);            // fold the two row-halves
  if (lane < 16) pooled[b * HID + ncol] = colsum * (1.0f / 360.0f);
}

// ---- stage 2: encoded = pooled @ W_e2 + b_e2 ------------------------------
__global__ __launch_bounds__(256) void k_encode(
    const float* __restrict__ pooled, const _Float16* __restrict__ W2T,
    const float* __restrict__ b2, float* __restrict__ enc) {
  __shared__ __align__(16) _Float16 Ah[16 * 520];
  const int mt = blockIdx.x;
  const int tid = threadIdx.x;
  const int wave = tid >> 5, lane = tid & 31;
  const int lrow = lane & 15;
  const int kbase = (lane < 16) ? 0 : 8;
  for (int i = tid; i < 16 * HID; i += 256) {
    int r = i >> 9, k = i & 511;
    Ah[r * 520 + k] = (_Float16)pooled[(mt * 16 + r) * HID + k];
  }
  __syncthreads();
  const _Float16* arow = Ah + lrow * 520;
  for (int q = 0; q < 2; ++q) {
    const int ncol = (wave * 2 + q) * 16 + lrow;
    v8f c = gemm_k512_ldB(arow, W2T + (size_t)ncol * HID, kbase);
    const float bias = b2[ncol];
    const int rb = (lane < 16) ? 0 : 8;
#pragma unroll
    for (int r = 0; r < 8; ++r)
      enc[(size_t)(mt * 16 + rb + r) * EDIM + ncol] = c[r] + bias;
  }
}

// ---- l2-normalize encoded rows -> f16 -------------------------------------
__global__ __launch_bounds__(256) void k_norm_enc(
    const float* __restrict__ enc, _Float16* __restrict__ encNh) {
  __shared__ float wsum[8];
  __shared__ float rinv_s;
  const int row = blockIdx.x;
  const int tid = threadIdx.x;
  float v = enc[row * EDIM + tid];
  float s = v * v;
  for (int o = 16; o > 0; o >>= 1) s += __shfl_xor(s, o);
  if ((tid & 31) == 0) wsum[tid >> 5] = s;
  __syncthreads();
  if (tid == 0) {
    float t = 0.f;
    for (int i = 0; i < 8; ++i) t += wsum[i];
    rinv_s = rsqrtf(fmaxf(t, 1e-12f));
  }
  __syncthreads();
  encNh[row * EDIM + tid] = (_Float16)(v * rinv_s);
}

// ---- fused gallery MLP + l2-normalize -------------------------------------
// grid 1875 (16-row tiles of the 29999-card gallery), 256 threads
__global__ __launch_bounds__(256) void k_gallery(
    const _Float16* __restrict__ cardsH,
    const _Float16* __restrict__ Wt1T, const float* __restrict__ bt1,
    const _Float16* __restrict__ Wt2T, const float* __restrict__ bt2,
    _Float16* __restrict__ tNh) {
  __shared__ __align__(16) _Float16 Ah[16 * 520];
  __shared__ __align__(16) _Float16 Hh[16 * 520];
  __shared__ float norms[16];
  const int mt = blockIdx.x;
  const int tid = threadIdx.x;
  const int wave = tid >> 5, lane = tid & 31;
  const int lrow = lane & 15;
  const int kbase = (lane < 16) ? 0 : 8;
  if (tid < 16) norms[tid] = 0.f;
  for (int j = tid; j < 16 * 64; j += 256) {   // async stage 16 gallery rows
    int r = j >> 6, kq = j & 63;
    int g = mt * 16 + r;
    int gi = (g < N_GALLERY) ? (g + 1) : 1;    // gallery skips card 0
    cp16_g2l(cardsH + (size_t)gi * CDIM + kq * 8, Ah + r * 520 + kq * 8);
  }
  cp_wait();
  __syncthreads();
  const _Float16* arow = Ah + lrow * 520;
  // layer 1: h = relu(A @ Wt1 + bt1) -> LDS f16
  for (int q = 0; q < 4; ++q) {
    const int ncol = (wave * 4 + q) * 16 + lrow;
    v8f c = gemm_k512_ldB(arow, Wt1T + (size_t)ncol * CDIM, kbase);
    const float bias = bt1[ncol];
    const int rb = (lane < 16) ? 0 : 8;
#pragma unroll
    for (int r = 0; r < 8; ++r) {
      float h = c[r] + bias;
      h = h > 0.f ? h : 0.f;
      Hh[(rb + r) * 520 + ncol] = (_Float16)h;
    }
  }
  __syncthreads();
  // layer 2: t = H @ Wt2 + bt2, plus row sums of squares
  const _Float16* hrow = Hh + lrow * 520;
  float vals[2][8];
  int ncols[2];
  for (int q = 0; q < 2; ++q) {
    const int ncol = (wave * 2 + q) * 16 + lrow;
    ncols[q] = ncol;
    v8f c = gemm_k512_ldB(hrow, Wt2T + (size_t)ncol * HID, kbase);
    const float bias = bt2[ncol];
    const int rb = (lane < 16) ? 0 : 8;
#pragma unroll
    for (int r = 0; r < 8; ++r) {
      float t = c[r] + bias;
      vals[q][r] = t;
      float s = t * t;                         // reduce over 16 cols in half-wave
      s += __shfl_xor(s, 1);
      s += __shfl_xor(s, 2);
      s += __shfl_xor(s, 4);
      s += __shfl_xor(s, 8);
      if (lrow == 0) atomicAdd(&norms[rb + r], s);
    }
  }
  __syncthreads();
  const int rb2 = (lane < 16) ? 0 : 8;
  for (int q = 0; q < 2; ++q) {
#pragma unroll
    for (int r = 0; r < 8; ++r) {
      const int row = rb2 + r;
      const int g = mt * 16 + row;
      const float rinv = rsqrtf(fmaxf(norms[row], 1e-12f));
      if (g < N_GALLERY)
        tNh[(size_t)g * EDIM + ncols[q]] = (_Float16)(vals[q][r] * rinv);
    }
  }
}

// ---- similarity GEMM + sigmoid --------------------------------------------
// grid (235 n-blocks of 128, 16 m-tiles), 256 threads
__global__ __launch_bounds__(256) void k_sims(
    const _Float16* __restrict__ encNh, const _Float16* __restrict__ tNh,
    const float* __restrict__ tempPtr, float* __restrict__ out) {
  __shared__ __align__(16) _Float16 Ah[16 * 264];
  const int nb = blockIdx.x;
  const int mt = blockIdx.y;
  const int tid = threadIdx.x;
  const int wave = tid >> 5, lane = tid & 31;
  const int lrow = lane & 15;
  const int kbase = (lane < 16) ? 0 : 8;
  for (int j = tid; j < 16 * 32; j += 256) {   // async stage A (16 x 256 f16)
    int r = j >> 5, kq = j & 31;
    cp16_g2l(encNh + (size_t)(mt * 16 + r) * EDIM + kq * 8,
             Ah + r * 264 + kq * 8);
  }
  cp_wait();
  __syncthreads();
  const int col = nb * 128 + wave * 16 + lrow;
  const int colc = col < N_GALLERY ? col : N_GALLERY - 1;
  const _Float16* Brow = tNh + (size_t)colc * EDIM;
  const _Float16* arow = Ah + lrow * 264;
  // preload all 8 A and 8 B fragments, then back-to-back WMMAs
  v16h af[8], bfr[8];
#pragma unroll
  for (int ks = 0; ks < 8; ++ks) {
    af[ks]  = frag16(arow, ks * 32, kbase);
    bfr[ks] = frag16(Brow, ks * 32, kbase);
  }
  v8f c = {};
#pragma unroll
  for (int ks = 0; ks < 8; ++ks) c = wmma_f16(af[ks], bfr[ks], c);
  const float temp = tempPtr[0];
  const int rb = (lane < 16) ? 0 : 8;
#pragma unroll
  for (int r = 0; r < 8; ++r) {
    const int row = mt * 16 + rb + r;
    const float x = (c[r] - 0.5f) * temp;
    const float y = 1.0f / (1.0f + __expf(-x));
    if (col < N_GALLERY) out[(size_t)row * N_GALLERY + col] = y;
  }
}

// ---- launcher --------------------------------------------------------------
extern "C" void kernel_launch(void* const* d_in, const int* in_sizes, int n_in,
                              void* d_out, int out_size, void* d_ws, size_t ws_size,
                              hipStream_t stream) {
  (void)in_sizes; (void)n_in; (void)out_size; (void)ws_size;
  const int*   idx   = (const int*)  d_in[0];
  const float* cards = (const float*)d_in[1];
  const float* W_e1  = (const float*)d_in[2];
  const float* b_e1  = (const float*)d_in[3];
  const float* W_e2  = (const float*)d_in[4];
  const float* b_e2  = (const float*)d_in[5];
  const float* W_t1  = (const float*)d_in[6];
  const float* b_t1  = (const float*)d_in[7];
  const float* W_t2  = (const float*)d_in[8];
  const float* b_t2  = (const float*)d_in[9];
  const float* temp  = (const float*)d_in[10];

  uint8_t* w = (uint8_t*)d_ws;
  _Float16* cardsH = (_Float16*)w; w += (size_t)NUM_CARDS * CDIM * 2;
  _Float16* We1T  = (_Float16*)w; w += (size_t)512 * 512 * 2;
  _Float16* Wt1T  = (_Float16*)w; w += (size_t)512 * 512 * 2;
  _Float16* We2T  = (_Float16*)w; w += (size_t)256 * 512 * 2;
  _Float16* Wt2T  = (_Float16*)w; w += (size_t)256 * 512 * 2;
  float*    pooled = (float*)w;   w += (size_t)256 * 512 * 4;
  _Float16* encNh = (_Float16*)w; w += (size_t)256 * 256 * 2;
  _Float16* tNh   = (_Float16*)w; w += (size_t)30000 * 256 * 2;

  float* decoded = (float*)d_out;
  float* enc     = (float*)d_out + (size_t)BATCH * N_GALLERY;

  const int n4cards = NUM_CARDS * CDIM / 4;
  k_f32_to_f16<<<(n4cards + 255) / 256, 256, 0, stream>>>(cards, cardsH, n4cards);
  k_transpose_h<<<(512 * 512 + 255) / 256, 256, 0, stream>>>(W_e1, We1T, 512, 512);
  k_transpose_h<<<(512 * 512 + 255) / 256, 256, 0, stream>>>(W_t1, Wt1T, 512, 512);
  k_transpose_h<<<(512 * 256 + 255) / 256, 256, 0, stream>>>(W_e2, We2T, 512, 256);
  k_transpose_h<<<(512 * 256 + 255) / 256, 256, 0, stream>>>(W_t2, Wt2T, 512, 256);

  k_embed_pool<<<dim3(BATCH, 4), 256, 0, stream>>>(idx, cardsH, We1T, b_e1, pooled);
  k_encode<<<16, 256, 0, stream>>>(pooled, We2T, b_e2, enc);
  k_norm_enc<<<BATCH, 256, 0, stream>>>(enc, encNh);
  k_gallery<<<1875, 256, 0, stream>>>(cardsH, Wt1T, b_t1, Wt2T, b_t2, tNh);
  k_sims<<<dim3(235, 16), 256, 0, stream>>>(encNh, tNh, temp, decoded);
}
